// PredDecoupleHead_13048110645921
// MI455X (gfx1250) — matrix operand order repaired and checked by reading, not compiled, not run on hardware
//
#include <hip/hip_runtime.h>

typedef __attribute__((ext_vector_type(16))) _Float16 v16h;
typedef __attribute__((ext_vector_type(8)))  _Float16 v8h;
typedef __attribute__((ext_vector_type(8)))  float    v8f;

#define NB 2
#define CIN 256
#define HH0 100
#define WW0 152
#define HWA0 (HH0*WW0)
#define HH1 50
#define WW1 76
#define HWA1 (HH1*WW1)
#define MCLS 720
#define MTILES_CLS 45
#define KDYN 2304
#define MDYN_PAD 48

#if defined(__has_builtin)
#if __has_builtin(__builtin_amdgcn_sched_barrier)
#define SCHED_BARRIER() __builtin_amdgcn_sched_barrier(0)
#endif
#endif
#ifndef SCHED_BARRIER
#define SCHED_BARRIER()
#endif

// ---------------------------------------------------------------- helpers
__device__ __forceinline__ float bilin_f(const float* __restrict__ m, int Wd, int Hd,
                                         float x, float y) {
  x = fminf(fmaxf(x, 0.f), (float)(Wd - 1));
  y = fminf(fmaxf(y, 0.f), (float)(Hd - 1));
  float xf = floorf(x), yf = floorf(y);
  float fx = x - xf, fy = y - yf;
  int x0 = (int)xf, y0 = (int)yf;
  int x1 = min(x0 + 1, Wd - 1), y1 = min(y0 + 1, Hd - 1);
  float v00 = m[y0 * Wd + x0], v01 = m[y0 * Wd + x1];
  float v10 = m[y1 * Wd + x0], v11 = m[y1 * Wd + x1];
  return v00 * (1.f - fx) * (1.f - fy) + v01 * fx * (1.f - fy) +
         v10 * (1.f - fx) * fy + v11 * fx * fy;
}

__device__ __forceinline__ float bilin_h(const _Float16* __restrict__ m, int Wd, int Hd,
                                         float x, float y) {
  x = fminf(fmaxf(x, 0.f), (float)(Wd - 1));
  y = fminf(fmaxf(y, 0.f), (float)(Hd - 1));
  float xf = floorf(x), yf = floorf(y);
  float fx = x - xf, fy = y - yf;
  int x0 = (int)xf, y0 = (int)yf;
  int x1 = min(x0 + 1, Wd - 1), y1 = min(y0 + 1, Hd - 1);
  float v00 = (float)m[y0 * Wd + x0], v01 = (float)m[y0 * Wd + x1];
  float v10 = (float)m[y1 * Wd + x0], v11 = (float)m[y1 * Wd + x1];
  return v00 * (1.f - fx) * (1.f - fy) + v01 * fx * (1.f - fy) +
         v10 * (1.f - fx) * fy + v11 * fx * fy;
}

// Load 16x32 f16 A-fragment: lane row = `row`, halves hold K = 16*half+0..15
// within each 32-wide k-step -> two 16B loads per step.
template <int KROW>
__device__ __forceinline__ void load_afrags(const _Float16* __restrict__ wh,
                                            int row, int half, v16h (&A)[8]) {
  const _Float16* wrow = wh + (size_t)row * KROW;
#pragma unroll
  for (int ks = 0; ks < 8; ++ks) {
    v8h lo = *(const v8h*)(wrow + ks * 32 + half * 16);
    v8h hi = *(const v8h*)(wrow + ks * 32 + half * 16 + 8);
#pragma unroll
    for (int i = 0; i < 8; ++i) { A[ks][i] = lo[i]; A[ks][i + 8] = hi[i]; }
  }
}

__device__ __forceinline__ v8f wmma8(const v16h (&A)[8], const v16h (&B)[8]) {
  v8f acc = {};
#pragma unroll
  for (int ks = 0; ks < 8; ++ks)
    acc = __builtin_amdgcn_wmma_f32_16x16x32_f16(false, A[ks], false, B[ks],
                                                 (short)0, acc, false, false);
  return acc;
}

// ------------------------------------------------- weight packing (f32->f16)
// combw[48][2304] with K ordered t*256+c (tap-major) so each 32-wide k-step
// has a uniform tap. Rows 0..33 = dyn_w, 34..37 = reg_w at center tap only
// (fused 1x1), 38..47 = zero pad.
__global__ void pack_weights(const float* __restrict__ dyn_w,
                             const float* __restrict__ reg_w,
                             const float* __restrict__ cls_w,
                             _Float16* __restrict__ combw,
                             _Float16* __restrict__ clsw) {
  int idx = blockIdx.x * blockDim.x + threadIdx.x;
  if (idx < MDYN_PAD * KDYN) {
    int o = idx / KDYN, r = idx % KDYN;
    int t = r / CIN, c = r % CIN;
    float v = 0.f;
    if (o < 34) v = dyn_w[((size_t)o * CIN + c) * 9 + t];
    else if (o < 38 && t == 4) v = reg_w[(size_t)(o - 34) * CIN + c];
    combw[idx] = (_Float16)v;
  }
  if (idx < MCLS * CIN) clsw[idx] = (_Float16)cls_w[idx];
}

// ------------------------------------------------- cls 1x1 GEMM (K=256, M=720)
// One wave owns a 16-pixel tile; the 8 B fragments (64 VGPRs) stay in
// registers. Per M-tile: all 16 A b128-loads issue first (sched barrier pins
// them), then 8 wmmas — waits stagger so later loads hide behind early wmmas.
template <int HW>
__global__ __launch_bounds__(128, 1) void cls_gemm(const float* __restrict__ feat,
                                                   const _Float16* __restrict__ wh,
                                                   _Float16* __restrict__ omap) {
  int wave = threadIdx.x >> 5;
  int lane = threadIdx.x & 31;
  int n = blockIdx.y;
  int pix0 = (blockIdx.x * 4 + wave) * 16;
  if (pix0 >= HW) return;                 // wave-uniform
  int half = lane >> 4;
  int pcol = lane & 15;
  int pix = min(pix0 + pcol, HW - 1);     // clamp; store is guarded
  int pixs = pix0 + pcol;
  const float* fb = feat + (size_t)n * CIN * HW + pix;

  v16h B[8];
#pragma unroll
  for (int ks = 0; ks < 8; ++ks) {
    int cbase = ks * 32 + half * 16;      // B: lanes 0-15 K=0..15, 16-31 K=16..31
#pragma unroll
    for (int i = 0; i < 16; ++i)
      B[ks][i] = (_Float16)fb[(size_t)(cbase + i) * HW];
  }

#pragma unroll 1
  for (int mt = 0; mt < MTILES_CLS; ++mt) {
    v16h A[8];
    load_afrags<CIN>(wh, mt * 16 + pcol, half, A);
    SCHED_BARRIER();                      // keep all 16 loads ahead of wmmas
    v8f acc = wmma8(A, B);
    if (pixs < HW) {
      _Float16* op = omap + ((size_t)n * MCLS + mt * 16 + half * 8) * HW + pixs;
#pragma unroll
      for (int r = 0; r < 8; ++r) op[(size_t)r * HW] = (_Float16)acc[r];
    }
  }
}

// ---------------- dyn 3x3 conv GEMM (K=2304 tap-major, M=48 incl. fused reg 1x1)
// Tap t is uniform per 32-wide k-step. Software-pipelined 2 deep: raw B floats
// and the 3 A fragments for step kk+1 prefetch while step kk's wmmas run.
__device__ __forceinline__ void load_am3(const _Float16* __restrict__ wh,
                                         int pcol, int half, int k0, v16h (&Am)[3]) {
#pragma unroll
  for (int mt = 0; mt < 3; ++mt) {
    const _Float16* wrow = wh + (size_t)(mt * 16 + pcol) * KDYN + k0;
    v8h lo = *(const v8h*)(wrow + half * 16);
    v8h hi = *(const v8h*)(wrow + half * 16 + 8);
#pragma unroll
    for (int i = 0; i < 8; ++i) { Am[mt][i] = lo[i]; Am[mt][i + 8] = hi[i]; }
  }
}

template <int H, int W>
__device__ __forceinline__ void dyn_step(const float* __restrict__ fb, int h, int w,
                                         int half, int kk,
                                         const float*& fp, float& maskf,
                                         int& k0, int& cb) {
  int t = kk >> 3;
  int cs = (kk & 7) << 5;
  int dh = t / 3 - 1, dw = t % 3 - 1;
  int hh = h + dh, ww = w + dw;
  maskf = (float)((hh >= 0) & (hh < H) & (ww >= 0) & (ww < W));
  fp = fb + min(max(hh, 0), H - 1) * W + min(max(ww, 0), W - 1);
  k0 = t * CIN + cs;
  cb = cs + half * 16;
}

template <int HW>
__device__ __forceinline__ void load_raw16(const float* __restrict__ fp, int cb,
                                           float (&fr)[16]) {
#pragma unroll
  for (int i = 0; i < 16; ++i) fr[i] = fp[(size_t)(cb + i) * HW];
}

__device__ __forceinline__ v16h cvt_b16(const float (&fr)[16], float maskf) {
  v16h B;
#pragma unroll
  for (int i = 0; i < 16; ++i) B[i] = (_Float16)(fr[i] * maskf);
  return B;
}

__device__ __forceinline__ void wmma3(const v16h (&Am)[3], v16h B,
                                      v8f& a0, v8f& a1, v8f& a2) {
  a0 = __builtin_amdgcn_wmma_f32_16x16x32_f16(false, Am[0], false, B, (short)0,
                                              a0, false, false);
  a1 = __builtin_amdgcn_wmma_f32_16x16x32_f16(false, Am[1], false, B, (short)0,
                                              a1, false, false);
  a2 = __builtin_amdgcn_wmma_f32_16x16x32_f16(false, Am[2], false, B, (short)0,
                                              a2, false, false);
}

template <int H, int W>
__global__ __launch_bounds__(128, 1) void dyn_gemm(const float* __restrict__ feat,
                                                   const _Float16* __restrict__ wh,
                                                   const float* __restrict__ dyn_b,
                                                   float* __restrict__ predreg) {
  constexpr int HW = H * W;
  int wave = threadIdx.x >> 5;
  int lane = threadIdx.x & 31;
  int n = blockIdx.y;
  int pix0 = (blockIdx.x * 4 + wave) * 16;
  if (pix0 >= HW) return;
  int half = lane >> 4;
  int pcol = lane & 15;
  int pix = min(pix0 + pcol, HW - 1);
  int pixs = pix0 + pcol;
  int h = pix / W, w = pix - h * W;
  const float* fb = feat + (size_t)n * CIN * HW;

  v8f acc0 = {}, acc1 = {}, acc2 = {};

  const float* fp0; float mk0; int k00, cb0;
  dyn_step<H, W>(fb, h, w, half, 0, fp0, mk0, k00, cb0);
  float fr0[16];
  load_raw16<HW>(fp0, cb0, fr0);
  v16h Am0[3];
  load_am3(wh, pcol, half, k00, Am0);

#pragma unroll 1
  for (int kk = 0; kk < 72; kk += 2) {
    // prefetch kk+1 (always valid: 72 is even)
    const float* fp1; float mk1; int k01, cb1;
    dyn_step<H, W>(fb, h, w, half, kk + 1, fp1, mk1, k01, cb1);
    float fr1[16];
    load_raw16<HW>(fp1, cb1, fr1);
    v16h Am1[3];
    load_am3(wh, pcol, half, k01, Am1);
    v16h B0 = cvt_b16(fr0, mk0);
    SCHED_BARRIER();                      // pin prefetch loads above wmmas
    wmma3(Am0, B0, acc0, acc1, acc2);

    if (kk + 2 < 72) {                    // prefetch kk+2
      dyn_step<H, W>(fb, h, w, half, kk + 2, fp0, mk0, k00, cb0);
      load_raw16<HW>(fp0, cb0, fr0);
      load_am3(wh, pcol, half, k00, Am0);
    }
    v16h B1 = cvt_b16(fr1, mk1);
    SCHED_BARRIER();
    wmma3(Am1, B1, acc0, acc1, acc2);
  }

  if (pixs < HW) {
#pragma unroll
    for (int mt = 0; mt < 3; ++mt) {
      v8f acc = (mt == 0 ? acc0 : (mt == 1 ? acc1 : acc2));
      float* op = predreg + ((size_t)n * MDYN_PAD + mt * 16 + half * 8) * HW + pixs;
#pragma unroll
      for (int r = 0; r < 8; ++r) {
        int o = mt * 16 + r + half * 8;
        if (o < 38) op[(size_t)r * HW] = acc[r] + (o < 34 ? dyn_b[o] : 0.f);
      }
    }
  }
}

// ----------------- assemble: coarse box, bound points, reg_box, sem points
template <int H, int W, int H_lo, int W_lo, int USE_ATTN>
__global__ void assemble_kernel(const float* __restrict__ predreg,
                                const float* __restrict__ predreg_lo,
                                const float* __restrict__ anchor,
                                float* __restrict__ out_coarse,
                                float* __restrict__ out_regbox,
                                float* __restrict__ sempt) {
  constexpr int HW = H * W;
  constexpr int HW_lo = H_lo * W_lo;
  int idx = blockIdx.x * blockDim.x + threadIdx.x;
  if (idx >= NB * HW) return;
  int n = idx / HW, pix = idx % HW;
  const float* pr = predreg + (size_t)n * MDYN_PAD * HW + pix;
#define PRC(c) pr[(size_t)(c) * HW]
  float c0 = anchor[(size_t)(n * 4 + 0) * HW + pix] + PRC(0);
  float c1 = anchor[(size_t)(n * 4 + 1) * HW + pix] + PRC(1);
  float c2 = anchor[(size_t)(n * 4 + 2) * HW + pix] + PRC(2);
  float c3 = anchor[(size_t)(n * 4 + 3) * HW + pix] + PRC(3);
  out_coarse[(size_t)(n * 4 + 0) * HW + pix] = c0;
  out_coarse[(size_t)(n * 4 + 1) * HW + pix] = c1;
  out_coarse[(size_t)(n * 4 + 2) * HW + pix] = c2;
  out_coarse[(size_t)(n * 4 + 3) * HW + pix] = c3;

  float cx = 0.5f * (c0 + c2), cy = 0.5f * (c1 + c3);
  float th0 = (c3 - c1) * 0.25f;   // /2 * RATIO(0.5)
  float th1 = (c2 - c0) * 0.25f;
  float thr[4] = {th0, th1, th0, th1};
  float b[4];
#pragma unroll
  for (int i = 0; i < 4; ++i) {
    float bo = PRC(4 + i);
    float extra = fmaxf(bo - thr[i], 0.f) + fminf(bo + thr[i], 0.f);
    b[i] = bo - extra;
  }
  // bound points: P0=(x1, cy+b0) P1=(cx+b1, y1) P2=(x2, cy+b2) P3=(cx+b3, y2)
  float bx[4] = {c0, cx + b[1], c2, cx + b[3]};
  float by[4] = {cy + b[0], c1, cy + b[2], c3};

  const float* rm = predreg + ((size_t)n * MDYN_PAD + 34) * HW;
  float regv[4];
#pragma unroll
  for (int p = 0; p < 4; ++p) {
    float hi = bilin_f(rm + (size_t)p * HW, W, H, bx[p], by[p]);
    if (USE_ATTN) {
      const float* rml = predreg_lo + ((size_t)n * MDYN_PAD + 34) * HW_lo;
      float lo = 0.5f * bilin_f(rml + (size_t)p * HW_lo, W_lo, H_lo,
                                bx[p] * 2.f, by[p] * 2.f);
      float a0 = PRC(26 + 2 * p), a1 = PRC(27 + 2 * p);
      float mx = fmaxf(a0, a1);
      float e0 = __expf(a0 - mx), e1 = __expf(a1 - mx);
      float w1 = e1 / (e0 + e1);
      hi = lo * (1.f - w1) + hi * w1;
    }
    regv[p] = hi;
  }
  // bound_pt_loc == coarse
  out_regbox[(size_t)(n * 4 + 0) * HW + pix] = c0 + regv[0];
  out_regbox[(size_t)(n * 4 + 1) * HW + pix] = c1 + regv[1];
  out_regbox[(size_t)(n * 4 + 2) * HW + pix] = c2 + regv[2];
  out_regbox[(size_t)(n * 4 + 3) * HW + pix] = c3 + regv[3];

  float X[3] = {c0, cx, c2}, Y[3] = {c1, cy, c3};
#pragma unroll
  for (int p = 0; p < 9; ++p) {
    int ix = p / 3, iy = p % 3;
    sempt[((size_t)n * 18 + 2 * p) * HW + pix] = X[ix] + PRC(8 + 2 * p);
    sempt[((size_t)n * 18 + 2 * p + 1) * HW + pix] = Y[iy] + PRC(9 + 2 * p);
  }
#undef PRC
}

// --------------------------- 9-point class gather: sum_p bilinear + bias
template <int H, int W>
__global__ void cls_gather(const _Float16* __restrict__ clsmap,
                           const float* __restrict__ sempt,
                           const float* __restrict__ cls_b,
                           float* __restrict__ out) {
  constexpr int HW = H * W;
  int idx = blockIdx.x * blockDim.x + threadIdx.x;
  if (idx >= NB * 80 * HW) return;
  int pix = idx % HW;
  int t = idx / HW;
  int cc = t % 80;
  int n = t / 80;
  float acc = cls_b[cc];
#pragma unroll
  for (int p = 0; p < 9; ++p) {
    float sx = sempt[((size_t)n * 18 + 2 * p) * HW + pix];
    float sy = sempt[((size_t)n * 18 + 2 * p + 1) * HW + pix];
    acc += bilin_h(clsmap + (size_t)(n * MCLS + p * 80 + cc) * HW, W, H, sx, sy);
  }
  out[idx] = acc;
}

// ================================================================= launch
extern "C" void kernel_launch(void* const* d_in, const int* in_sizes, int n_in,
                              void* d_out, int out_size, void* d_ws, size_t ws_size,
                              hipStream_t stream) {
  (void)in_sizes; (void)n_in; (void)out_size; (void)ws_size;
  const float* reg_feat0 = (const float*)d_in[0];
  const float* reg_feat1 = (const float*)d_in[1];
  const float* cls_feat0 = (const float*)d_in[2];
  const float* cls_feat1 = (const float*)d_in[3];
  const float* anchor0   = (const float*)d_in[4];
  const float* anchor1   = (const float*)d_in[5];
  const float* dyn_w     = (const float*)d_in[6];
  const float* dyn_b     = (const float*)d_in[7];
  const float* reg_w     = (const float*)d_in[8];
  const float* cls_w     = (const float*)d_in[9];
  const float* cls_b     = (const float*)d_in[10];
  float* out = (float*)d_out;

  // workspace layout (256B aligned segments)
  char* ws = (char*)d_ws;
  size_t off = 0;
  auto take = [&](size_t bytes) {
    size_t o = off;
    off = (off + bytes + 255) & ~(size_t)255;
    return o;
  };
  _Float16* clsmap0 = (_Float16*)(ws + take((size_t)NB * MCLS * HWA0 * 2));
  _Float16* clsmap1 = (_Float16*)(ws + take((size_t)NB * MCLS * HWA1 * 2));
  float* predreg0   = (float*)(ws + take((size_t)NB * MDYN_PAD * HWA0 * 4));
  float* predreg1   = (float*)(ws + take((size_t)NB * MDYN_PAD * HWA1 * 4));
  float* sempt0     = (float*)(ws + take((size_t)NB * 18 * HWA0 * 4));
  float* sempt1     = (float*)(ws + take((size_t)NB * 18 * HWA1 * 4));
  _Float16* combw   = (_Float16*)(ws + take((size_t)MDYN_PAD * KDYN * 2));
  _Float16* clsw    = (_Float16*)(ws + take((size_t)MCLS * CIN * 2));

  // output offsets (return order: cls0, cls1, coarse0, coarse1, reg0, reg1)
  float* out_cls0    = out;
  float* out_cls1    = out_cls0 + (size_t)NB * 80 * HWA0;
  float* out_coarse0 = out_cls1 + (size_t)NB * 80 * HWA1;
  float* out_coarse1 = out_coarse0 + (size_t)NB * 4 * HWA0;
  float* out_reg0    = out_coarse1 + (size_t)NB * 4 * HWA1;
  float* out_reg1    = out_reg0 + (size_t)NB * 4 * HWA0;

  // 1) pack weights to f16
  {
    int total = MCLS * CIN; // 184320 > 48*2304
    pack_weights<<<(total + 255) / 256, 256, 0, stream>>>(dyn_w, reg_w, cls_w,
                                                          combw, clsw);
  }
  // 2) cls 1x1 GEMMs
  {
    dim3 g0((((HWA0 + 15) / 16) + 3) / 4, NB);
    cls_gemm<HWA0><<<g0, 128, 0, stream>>>(cls_feat0, clsw, clsmap0);
    dim3 g1((((HWA1 + 15) / 16) + 3) / 4, NB);
    cls_gemm<HWA1><<<g1, 128, 0, stream>>>(cls_feat1, clsw, clsmap1);
  }
  // 3) dyn 3x3 (+ fused reg 1x1) GEMMs
  {
    dim3 g0((((HWA0 + 15) / 16) + 3) / 4, NB);
    dyn_gemm<HH0, WW0><<<g0, 128, 0, stream>>>(reg_feat0, combw, dyn_b, predreg0);
    dim3 g1((((HWA1 + 15) / 16) + 3) / 4, NB);
    dyn_gemm<HH1, WW1><<<g1, 128, 0, stream>>>(reg_feat1, combw, dyn_b, predreg1);
  }
  // 4) assemble (level 1 samples level-0 reg map -> ordered on stream)
  {
    int t0 = NB * HWA0;
    assemble_kernel<HH0, WW0, HH0, WW0, 0><<<(t0 + 255) / 256, 256, 0, stream>>>(
        predreg0, predreg0, anchor0, out_coarse0, out_reg0, sempt0);
    int t1 = NB * HWA1;
    assemble_kernel<HH1, WW1, HH0, WW0, 1><<<(t1 + 255) / 256, 256, 0, stream>>>(
        predreg1, predreg0, anchor1, out_coarse1, out_reg1, sempt1);
  }
  // 5) class gathers
  {
    int t0 = NB * 80 * HWA0;
    cls_gather<HH0, WW0><<<(t0 + 255) / 256, 256, 0, stream>>>(clsmap0, sempt0,
                                                               cls_b, out_cls0);
    int t1 = NB * 80 * HWA1;
    cls_gather<HH1, WW1><<<(t1 + 255) / 256, 256, 0, stream>>>(clsmap1, sempt1,
                                                               cls_b, out_cls1);
  }
}
